// GraphVAE_3624952398187
// MI455X (gfx1250) — matrix operand order repaired and checked by reading, not compiled, not run on hardware
//
#include <hip/hip_runtime.h>

#define NN 50000
#define NE 1600000
#define NG 256
#define DH 128
#define DL 64
#define MAXN 64

typedef __attribute__((ext_vector_type(16))) __bf16 v16bf;
typedef __attribute__((ext_vector_type(8)))  __bf16 v8bf;
typedef __attribute__((ext_vector_type(8)))  float  v8f;

__device__ __forceinline__ __bf16 f2bf(float f) {
  unsigned u = __builtin_bit_cast(unsigned, f);
  u += 0x7FFFu + ((u >> 16) & 1u);                // round-to-nearest-even
  unsigned short s = (unsigned short)(u >> 16);
  return __builtin_bit_cast(__bf16, s);
}

// ---------------- elementwise / memory kernels ----------------

__global__ void k_zero(float* __restrict__ p, long long n) {
  long long i = (long long)blockIdx.x * blockDim.x + threadIdx.x;
  if (i < n) p[i] = 0.f;
}

// agg[dst[e]][:] += h[src[e]][:]   (32 lanes per edge, float4 per lane)
__global__ void k_edge_agg(const float* __restrict__ h, const int* __restrict__ src,
                           const int* __restrict__ dst, float* __restrict__ agg) {
  long long t = (long long)blockIdx.x * blockDim.x + threadIdx.x;
  long long e = t >> 5;
  if (e >= NE) return;
  int lane = (int)(t & 31);
  int s = src[e], d = dst[e];
  float4 v = ((const float4*)(h + (long long)s * DH))[lane];
  float* ap = agg + (long long)d * DH + lane * 4;
  unsafeAtomicAdd(ap + 0, v.x);
  unsafeAtomicAdd(ap + 1, v.y);
  unsafeAtomicAdd(ap + 2, v.z);
  unsafeAtomicAdd(ap + 3, v.w);
}

// xin = bf16((1+eps)*h + agg)
__global__ void k_xin(const float* __restrict__ h, const float* __restrict__ agg,
                      const float* __restrict__ epsp, __bf16* __restrict__ xin, long long n) {
  long long i = (long long)blockIdx.x * blockDim.x + threadIdx.x;
  if (i >= n) return;
  float e1 = 1.0f + epsp[0];
  xin[i] = f2bf(e1 * h[i] + agg[i]);
}

// W[K x N] row-major fp32 -> Wt[N x K] bf16 (contiguous K per output column)
__global__ void k_wconvT(const float* __restrict__ W, __bf16* __restrict__ Wt, int K, int N) {
  long long i = (long long)blockIdx.x * blockDim.x + threadIdx.x;
  if (i >= (long long)K * N) return;
  int k = (int)(i / N), n = (int)(i % N);
  Wt[(long long)n * K + k] = f2bf(W[i]);
}

// per-feature sum / sumsq partials (128 features, block of 128 threads walks 256 rows)
__global__ void k_bn_stats(const float* __restrict__ m, float* __restrict__ stats, int rows) {
  int f = threadIdx.x;
  int r0 = blockIdx.x * 256;
  int r1 = r0 + 256; if (r1 > rows) r1 = rows;
  float s = 0.f, q = 0.f;
  for (int r = r0; r < r1; ++r) {
    float v = m[(long long)r * DH + f];
    s += v; q += v * v;
  }
  unsafeAtomicAdd(&stats[f], s);
  unsafeAtomicAdd(&stats[DH + f], q);
}

__global__ void k_bn_apply(const float* __restrict__ m, const float* __restrict__ stats,
                           const float* __restrict__ gamma, const float* __restrict__ beta,
                           float* __restrict__ hout, int rows) {
  long long i = (long long)blockIdx.x * blockDim.x + threadIdx.x;
  if (i >= (long long)rows * DH) return;
  int f = (int)(i & (DH - 1));
  float invn = 1.0f / (float)rows;
  float mean = stats[f] * invn;
  float var  = stats[DH + f] * invn - mean * mean;
  float inv  = rsqrtf(var + 1e-5f);
  float v = gamma[f] * (m[i] - mean) * inv + beta[f];
  hout[i] = fmaxf(v, 0.f);
}

// per-graph sums + counts (32 lanes per node row)
__global__ void k_pool(const float* __restrict__ h, const int* __restrict__ batch,
                       float* __restrict__ sums, float* __restrict__ cnt) {
  long long t = (long long)blockIdx.x * blockDim.x + threadIdx.x;
  if (t >= (long long)NN * 32) return;
  int r = (int)(t >> 5), lane = (int)(t & 31);
  int g = batch[r];
  float4 v = ((const float4*)(h + (long long)r * DH))[lane];
  float* sp = sums + (long long)g * DH + lane * 4;
  unsafeAtomicAdd(sp + 0, v.x);
  unsafeAtomicAdd(sp + 1, v.y);
  unsafeAtomicAdd(sp + 2, v.z);
  unsafeAtomicAdd(sp + 3, v.w);
  if (lane == 0) unsafeAtomicAdd(&cnt[g], 1.0f);
}

__global__ void k_pool_div(const float* __restrict__ sums, const float* __restrict__ cnt,
                           __bf16* __restrict__ hgb) {
  int i = blockIdx.x * blockDim.x + threadIdx.x;
  if (i >= NG * DH) return;
  int g = i / DH;
  float c = fmaxf(cnt[g], 1.0f);
  hgb[i] = f2bf(sums[i] / c);
}

// z = mu + noise * exp(0.5*logvar)  (bf16 for decoder GEMM)
__global__ void k_z(const float* __restrict__ mu, const float* __restrict__ lv,
                    const float* __restrict__ noise, __bf16* __restrict__ zb) {
  int i = blockIdx.x * blockDim.x + threadIdx.x;
  if (i >= NG * DL) return;
  zb[i] = f2bf(mu[i] + noise[i] * expf(0.5f * lv[i]));
}

// ---------------- WMMA GEMM ----------------
// C[M,N] = A[M,K](bf16,row-major) * Bt[N,K](bf16)^T + bias, optional relu.
// M and N are multiples of 16; K in {64,128} (compile-time -> fully unrolled).
// One wave computes a 16x32 tile (two 16x16 WMMA tiles sharing one A fragment).
template <int K, bool RELU>
__global__ void k_gemm(const __bf16* __restrict__ A, const __bf16* __restrict__ Bt,
                       const float* __restrict__ bias, float* __restrict__ Cf,
                       __bf16* __restrict__ Cb, int N) {
  int lane = threadIdx.x;                                // 0..31
  int r0   = blockIdx.x * 16;
  int pair = blockIdx.y * blockDim.y + threadIdx.y;      // 32-column group
  int n0   = pair * 32;
  int hs   = lane >> 4;                                  // half-select
  int m16  = lane & 15;
  long long row  = r0 + m16;                             // A fragment row
  long long col0 = n0 + m16;                             // left tile column
  long long col1 = col0 + 16;                            // right tile column

  v8f acc0 = {}, acc1 = {};
  #pragma unroll
  for (int k0 = 0; k0 < K; k0 += 32) {
    // A (16x32 bf16): lane = row M; K = 16 contiguous starting at k0 + hs*16
    const __bf16* ap = A + row * K + k0 + hs * 16;
    v8bf alo = *(const v8bf*)(ap);
    v8bf ahi = *(const v8bf*)(ap + 8);
    v16bf a = __builtin_shufflevector(alo, ahi, 0,1,2,3,4,5,6,7,8,9,10,11,12,13,14,15);

    // B (32x16 bf16): lane = col N; K = 16 contiguous starting at k0 + hs*16
    const __bf16* bp0 = Bt + col0 * K + k0 + hs * 16;
    v8bf b0l = *(const v8bf*)(bp0);
    v8bf b0h = *(const v8bf*)(bp0 + 8);
    v16bf b0 = __builtin_shufflevector(b0l, b0h, 0,1,2,3,4,5,6,7,8,9,10,11,12,13,14,15);

    const __bf16* bp1 = Bt + col1 * K + k0 + hs * 16;
    v8bf b1l = *(const v8bf*)(bp1);
    v8bf b1h = *(const v8bf*)(bp1 + 8);
    v16bf b1 = __builtin_shufflevector(b1l, b1h, 0,1,2,3,4,5,6,7,8,9,10,11,12,13,14,15);

    acc0 = __builtin_amdgcn_wmma_f32_16x16x32_bf16(false, a, false, b0, (short)0, acc0, false, false);
    acc1 = __builtin_amdgcn_wmma_f32_16x16x32_bf16(false, a, false, b1, (short)0, acc1, false, false);
  }

  float bv0 = bias ? bias[col0] : 0.f;
  float bv1 = bias ? bias[col1] : 0.f;
  #pragma unroll
  for (int i = 0; i < 8; ++i) {                          // VGPR i <-> M = i + hs*8
    long long r = r0 + i + hs * 8;
    float v0 = acc0[i] + bv0;
    float v1 = acc1[i] + bv1;
    if (RELU) { v0 = fmaxf(v0, 0.f); v1 = fmaxf(v1, 0.f); }
    if (Cf) { Cf[r * N + col0] = v0; Cf[r * N + col1] = v1; }
    if (Cb) { Cb[r * N + col0] = f2bf(v0); Cb[r * N + col1] = f2bf(v1); }
  }
}

// ---------------- host side ----------------

static inline void launch_zero(float* p, long long n, hipStream_t s) {
  k_zero<<<(unsigned)((n + 255) / 256), 256, 0, s>>>(p, n);
}

static inline void launch_gemm(const __bf16* A, const __bf16* Bt, const float* bias,
                               float* Cf, __bf16* Cb, int M, int N, int K, int relu,
                               hipStream_t s) {
  int pairs = N / 32;                  // all N here are multiples of 32
  int wy = pairs < 8 ? pairs : 8;
  dim3 grid(M / 16, pairs / wy);       // all M here are multiples of 16
  dim3 block(32, wy);
  if (K == 128) {
    if (relu) k_gemm<128, true ><<<grid, block, 0, s>>>(A, Bt, bias, Cf, Cb, N);
    else      k_gemm<128, false><<<grid, block, 0, s>>>(A, Bt, bias, Cf, Cb, N);
  } else {
    if (relu) k_gemm<64, true ><<<grid, block, 0, s>>>(A, Bt, bias, Cf, Cb, N);
    else      k_gemm<64, false><<<grid, block, 0, s>>>(A, Bt, bias, Cf, Cb, N);
  }
}

extern "C" void kernel_launch(void* const* d_in, const int* in_sizes, int n_in,
                              void* d_out, int out_size, void* d_ws, size_t ws_size,
                              hipStream_t stream) {
  (void)in_sizes; (void)n_in; (void)out_size; (void)ws_size;

  const float* x     = (const float*)d_in[0];
  const int*   eidx  = (const int*)  d_in[1];
  const int*   src   = eidx;
  const int*   dst   = eidx + NE;
  const int*   batch = (const int*)  d_in[2];
  const float* fc_mean_W   = (const float*)d_in[24];
  const float* fc_mean_b   = (const float*)d_in[25];
  const float* fc_logvar_W = (const float*)d_in[26];
  const float* fc_logvar_b = (const float*)d_in[27];
  const float* dec_W1 = (const float*)d_in[28];
  const float* dec_b1 = (const float*)d_in[29];
  const float* dec_W2 = (const float*)d_in[30];
  const float* dec_b2 = (const float*)d_in[31];
  const float* eps_noise = (const float*)d_in[32];

  // ---- carve workspace ----
  char* w = (char*)d_ws;
  auto carve = [&](size_t bytes) { void* p = (void*)w; w += (bytes + 255) & ~(size_t)255; return p; };
  float*  hbuf  = (float*) carve((size_t)NN * DH * 4);   // BN output per layer
  float*  aggm  = (float*) carve((size_t)NN * DH * 4);   // agg, then reused as MLP output m
  __bf16* xin   = (__bf16*)carve((size_t)NN * DH * 2);   // (1+eps)h+agg  (GEMM1 A)
  __bf16* tbuf  = (__bf16*)carve((size_t)NN * DH * 2);   // relu(xin@Wa)  (GEMM2 A)
  float*  stats = (float*) carve(2 * DH * 4);            // sum / sumsq
  float*  pools = (float*) carve((size_t)NG * DH * 4);
  float*  cnts  = (float*) carve((size_t)NG * 4);
  __bf16* hgb   = (__bf16*)carve((size_t)NG * DH * 2);
  __bf16* zb    = (__bf16*)carve((size_t)NG * DL * 2);
  __bf16* hidb  = (__bf16*)carve((size_t)NG * DH * 2);
  __bf16* wt    = (__bf16*)carve((size_t)4096 * DH * 2); // reused transposed bf16 weights

  // ---- output regions ----
  float* logits = (float*)d_out;                           // 256*64*64
  float* mu_o   = logits + (long long)NG * MAXN * MAXN;    // 256*64
  float* lv_o   = mu_o + (long long)NG * DL;               // 256*64

  const long long nElem = (long long)NN * DH;
  const float* hcur = x;

  // ---- 3 GIN layers ----
  for (int l = 0; l < 3; ++l) {
    const float* Wa  = (const float*)d_in[3 + 7 * l + 0];
    const float* ba  = (const float*)d_in[3 + 7 * l + 1];
    const float* Wb  = (const float*)d_in[3 + 7 * l + 2];
    const float* bb  = (const float*)d_in[3 + 7 * l + 3];
    const float* gam = (const float*)d_in[3 + 7 * l + 4];
    const float* bet = (const float*)d_in[3 + 7 * l + 5];
    const float* eps = (const float*)d_in[3 + 7 * l + 6];

    launch_zero(aggm, nElem, stream);
    {
      long long th = (long long)NE * 32;
      k_edge_agg<<<(unsigned)((th + 255) / 256), 256, 0, stream>>>(hcur, src, dst, aggm);
    }
    k_xin<<<(unsigned)((nElem + 255) / 256), 256, 0, stream>>>(hcur, aggm, eps, xin, nElem);

    k_wconvT<<<(DH * DH + 255) / 256, 256, 0, stream>>>(Wa, wt, DH, DH);
    launch_gemm(xin, wt, ba, nullptr, tbuf, NN, DH, DH, /*relu=*/1, stream);

    k_wconvT<<<(DH * DH + 255) / 256, 256, 0, stream>>>(Wb, wt, DH, DH);
    launch_gemm(tbuf, wt, bb, aggm, nullptr, NN, DH, DH, /*relu=*/0, stream);  // m -> aggm

    launch_zero(stats, 2 * DH, stream);
    k_bn_stats<<<(NN + 255) / 256, DH, 0, stream>>>(aggm, stats, NN);
    k_bn_apply<<<(unsigned)((nElem + 255) / 256), 256, 0, stream>>>(aggm, stats, gam, bet, hbuf, NN);
    hcur = hbuf;
  }

  // ---- global mean pool ----
  launch_zero(pools, (long long)NG * DH, stream);
  launch_zero(cnts, NG, stream);
  {
    long long th = (long long)NN * 32;
    k_pool<<<(unsigned)((th + 255) / 256), 256, 0, stream>>>(hcur, batch, pools, cnts);
  }
  k_pool_div<<<(NG * DH + 255) / 256, 256, 0, stream>>>(pools, cnts, hgb);

  // ---- mu / logvar heads ----
  k_wconvT<<<(DH * DL + 255) / 256, 256, 0, stream>>>(fc_mean_W, wt, DH, DL);
  launch_gemm(hgb, wt, fc_mean_b, mu_o, nullptr, NG, DL, DH, 0, stream);
  k_wconvT<<<(DH * DL + 255) / 256, 256, 0, stream>>>(fc_logvar_W, wt, DH, DL);
  launch_gemm(hgb, wt, fc_logvar_b, lv_o, nullptr, NG, DL, DH, 0, stream);

  // ---- reparameterize ----
  k_z<<<(NG * DL + 255) / 256, 256, 0, stream>>>(mu_o, lv_o, eps_noise, zb);

  // ---- decoder ----
  k_wconvT<<<(DL * DH + 255) / 256, 256, 0, stream>>>(dec_W1, wt, DL, DH);
  launch_gemm(zb, wt, dec_b1, nullptr, hidb, NG, DH, DL, /*relu=*/1, stream);
  k_wconvT<<<(DH * 4096 + 255) / 256, 256, 0, stream>>>(dec_W2, wt, DH, MAXN * MAXN);
  launch_gemm(hidb, wt, dec_b2, logits, nullptr, NG, MAXN * MAXN, DH, 0, stream);
}
